// DeepSeekAttention_730144440685
// MI455X (gfx1250) — matrix operand order repaired and checked by reading, not compiled, not run on hardware
//
#include <hip/hip_runtime.h>
#include <hip/hip_bf16.h>

typedef __bf16 bf16;
typedef __attribute__((ext_vector_type(16))) __bf16 v16bf;
typedef __attribute__((ext_vector_type(8)))  __bf16 v8bf;
typedef __attribute__((ext_vector_type(8)))  float  v8f;

#define T_SEQ  2048
#define C_DIM  2048
#define D_HEAD 128
#define NHEAD  16
#define NKV    4
#define WIN    256
#define NQKV   3072          // 2048 q + 512 k + 512 v
#define NEG_BIG (-1e30f)

// ---------------- helpers ----------------

__device__ __forceinline__ bf16 f2bf(float f) {
  unsigned u = __builtin_bit_cast(unsigned, f);
  u += 0x7FFFu + ((u >> 16) & 1u);                 // round-to-nearest-even
  unsigned short h = (unsigned short)(u >> 16);
  return __builtin_bit_cast(bf16, h);
}

__device__ __forceinline__ v8f zero8() {
  v8f z;
#pragma unroll
  for (int i = 0; i < 8; ++i) z[i] = 0.0f;
  return z;
}

__device__ __forceinline__ float redmax16(float v) {
  v = fmaxf(v, __shfl_xor(v, 1, 16));
  v = fmaxf(v, __shfl_xor(v, 2, 16));
  v = fmaxf(v, __shfl_xor(v, 4, 16));
  v = fmaxf(v, __shfl_xor(v, 8, 16));
  return v;
}
__device__ __forceinline__ float redsum16(float v) {
  v += __shfl_xor(v, 1, 16);
  v += __shfl_xor(v, 2, 16);
  v += __shfl_xor(v, 4, 16);
  v += __shfl_xor(v, 8, 16);
  return v;
}

// ---- WMMA fragment loaders (wave32, v_wmma_f32_16x16x32_bf16) ----
// A-frag (16 rows x 32 K, row-major source): lane L -> row L%16;
// lanes 0-15 hold K={0..7,16..23}, lanes 16-31 hold K={8..15,24..31}.
__device__ __forceinline__ v16bf load_a_frag(const bf16* A16, int lda, int kBase, int lane) {
  const int row  = lane & 15;
  const int koff = (lane >> 4) << 3;
  const bf16* p = A16 + (size_t)row * lda + kBase + koff;
  v8bf lo = *(const v8bf*)(p);
  v8bf hi = *(const v8bf*)(p + 16);
  v16bf f;
#pragma unroll
  for (int e = 0; e < 8; ++e) { f[e] = lo[e]; f[e + 8] = hi[e]; }
  return f;
}

// B-frag (32 K x 16 N), source given transposed: BT[n][k] row-major.
// lanes 0-15: K={0..7,16..23}; lanes 16-31: K={8..15,24..31} (matches A-frag).
__device__ __forceinline__ v16bf load_bt_frag(const bf16* BT, int ldk, int nBase, int kBase, int lane) {
  const int n = nBase + (lane & 15);
  const int k = kBase + ((lane >> 4) << 3);
  const bf16* p = BT + (size_t)n * ldk + k;
  v8bf lo = *(const v8bf*)(p);
  v8bf hi = *(const v8bf*)(p + 16);
  v16bf f;
#pragma unroll
  for (int e = 0; e < 8; ++e) { f[e] = lo[e]; f[e + 8] = hi[e]; }
  return f;
}

#define WMMA_BF16(a, b, c) \
  __builtin_amdgcn_wmma_f32_16x16x32_bf16(false, (a), false, (b), (short)0, (c), false, false)

// safe online-softmax update for one C-layout row element (16 lanes hold the 16 cols)
__device__ __forceinline__ void sm_update(float s0v, float s1v, float& m, float& l,
                                          float& alpha, float& p0, float& p1) {
  const float mx = redmax16(fmaxf(s0v, s1v));
  const float mn = fmaxf(m, mx);
  const float safe = (mn <= NEG_BIG * 0.5f) ? 0.f : mn;   // fully-masked step -> everything exp()->0
  alpha = __expf(m - safe);
  p0 = __expf(s0v - safe);
  p1 = __expf(s1v - safe);
  l  = l * alpha + redsum16(p0 + p1);
  m  = mn;
}

// ---------------- conversion / packing kernels ----------------

__global__ void cvt_f32_bf16_kernel(const float* __restrict__ in, bf16* __restrict__ out, int n) {
  int i = blockIdx.x * blockDim.x + threadIdx.x;
  if (i < n) out[i] = f2bf(in[i]);
}

// wqkvT[n][c] = Wcat[c][n], Wcat = [wq | wk | wv] along N (3072 cols)
__global__ void pack_wqkv_kernel(const float* __restrict__ wq, const float* __restrict__ wk,
                                 const float* __restrict__ wv, bf16* __restrict__ wt) {
  int i = blockIdx.x * blockDim.x + threadIdx.x;
  if (i >= NQKV * C_DIM) return;
  const int n = i >> 11;        // /2048
  const int c = i & (C_DIM - 1);
  float v;
  if (n < 2048)       v = wq[(size_t)c * 2048 + n];
  else if (n < 2560)  v = wk[(size_t)c * 512 + (n - 2048)];
  else                v = wv[(size_t)c * 512 + (n - 2560)];
  wt[i] = f2bf(v);
}

__global__ void transpose_wo_kernel(const float* __restrict__ wo, bf16* __restrict__ wt) {
  int i = blockIdx.x * blockDim.x + threadIdx.x;
  if (i >= C_DIM * C_DIM) return;
  const int n = i >> 11;
  const int c = i & (C_DIM - 1);
  wt[i] = f2bf(wo[(size_t)c * C_DIM + n]);
}

// ---------------- bf16 TN GEMM: C[M][N] = A[M][K] * BT[N][K]^T ----------------
// 4 waves / block = 4 quadrants of a 128x128 block tile; each wave owns a
// 64x64 tile (16 accumulators, 16 WMMAs per K=32 chunk -> 32 FLOP per L2 byte).

__global__ __launch_bounds__(128) void gemm_bf16_tn(const bf16* __restrict__ A,
                                                    const bf16* __restrict__ BT,
                                                    float* __restrict__ C,
                                                    int M, int N, int K) {
  const int lane = threadIdx.x & 31;
  const int wave = threadIdx.x >> 5;
  const int m0 = blockIdx.y * 128 + (wave >> 1) * 64;
  const int n0 = blockIdx.x * 128 + (wave & 1) * 64;

  v8f acc[4][4];
#pragma unroll
  for (int mi = 0; mi < 4; ++mi)
#pragma unroll
    for (int ni = 0; ni < 4; ++ni) acc[mi][ni] = zero8();

  for (int kb = 0; kb < K; kb += 32) {
    v16bf af[4], bfr[4];
#pragma unroll
    for (int mi = 0; mi < 4; ++mi)
      af[mi] = load_a_frag(A + (size_t)(m0 + 16 * mi) * K, K, kb, lane);
#pragma unroll
    for (int ni = 0; ni < 4; ++ni)
      bfr[ni] = load_bt_frag(BT, K, n0 + 16 * ni, kb, lane);
    if (kb + 32 < K) {   // pull next K-chunk into GL2 (global_prefetch_b8)
#pragma unroll
      for (int mi = 0; mi < 4; ++mi)
        __builtin_prefetch(A + (size_t)(m0 + 16 * mi + (lane & 15)) * K + kb + 32, 0, 1);
#pragma unroll
      for (int ni = 0; ni < 4; ++ni)
        __builtin_prefetch(BT + (size_t)(n0 + 16 * ni + (lane & 15)) * K + kb + 32, 0, 1);
    }
#pragma unroll
    for (int mi = 0; mi < 4; ++mi)
#pragma unroll
      for (int ni = 0; ni < 4; ++ni)
        acc[mi][ni] = WMMA_BF16(af[mi], bfr[ni], acc[mi][ni]);
  }

  const int colb = lane & 15;
  const int rb   = (lane >> 4) << 3;
#pragma unroll
  for (int mi = 0; mi < 4; ++mi)
#pragma unroll
    for (int ni = 0; ni < 4; ++ni)
#pragma unroll
      for (int i = 0; i < 8; ++i)
        C[(size_t)(m0 + 16 * mi + rb + i) * N + n0 + 16 * ni + colb] = acc[mi][ni][i];
}

// ---------------- RoPE + layout pack ----------------
// qkv[t][0..2047]=q, [2048..2559]=k, [2560..3071]=v
// q -> qb[h][t][d] (scaled by 1/sqrt(D)); k -> kbuf[kv][t][d]; v -> vtb[kv][d][t]

__global__ void rope_pack_kernel(const float* __restrict__ qkv, bf16* __restrict__ qb,
                                 bf16* __restrict__ kbuf, bf16* __restrict__ vtb) {
  const int t = blockIdx.x;
  const float* row = qkv + (size_t)t * NQKV;
  const float tf = (float)t;
  const float rscale = 0.08838834764831845f;       // 1/sqrt(128)
  const float lnb_d  = 0.14391156516f;             // ln(10000)/64
  for (int c = threadIdx.x; c < NQKV; c += blockDim.x) {
    if (c < 2048) {
      const int hh = c >> 7, d = c & 127, j = d & 63;
      const float inv = __expf(-(float)j * lnb_d);
      float sn, cs; sincosf(tf * inv, &sn, &cs);
      const float x0 = row[c];
      const float xr = (d < 64) ? -row[(hh << 7) + d + 64] : row[(hh << 7) + d - 64];
      qb[((size_t)hh * T_SEQ + t) * D_HEAD + d] = f2bf((x0 * cs + xr * sn) * rscale);
    } else if (c < 2560) {
      const int cc = c - 2048, hh = cc >> 7, d = cc & 127, j = d & 63;
      const float inv = __expf(-(float)j * lnb_d);
      float sn, cs; sincosf(tf * inv, &sn, &cs);
      const float x0 = row[c];
      const float xr = (d < 64) ? -row[2048 + (hh << 7) + d + 64] : row[2048 + (hh << 7) + d - 64];
      kbuf[((size_t)hh * T_SEQ + t) * D_HEAD + d] = f2bf(x0 * cs + xr * sn);
    } else {
      const int cc = c - 2560, hh = cc >> 7, d = cc & 127;
      vtb[((size_t)hh * D_HEAD + d) * T_SEQ + t] = f2bf(row[c]);
    }
  }
}

// ---------------- fused dual-softmax flash attention ----------------
// one wave per (16-query tile, head). Q pre-scaled. Causal + local-window softmax
// in one pass over keys; local range is always a subset of the causal range.

__global__ __launch_bounds__(32) void attn_fused_kernel(const bf16* __restrict__ qbuf,
                                                        const bf16* __restrict__ kbuf,
                                                        const bf16* __restrict__ vtbuf,
                                                        bf16* __restrict__ attn) {
  __shared__ bf16 ldsP[2][16][32];
  const int lane = threadIdx.x;
  const int q0 = blockIdx.x * 16;
  const int h  = blockIdx.y;
  const int kv = h >> 2;                    // repeat_interleave: kv head = h / 4
  const int colb = lane & 15;
  const int half = lane >> 4;

  const bf16* Q  = qbuf  + ((size_t)h  * T_SEQ + q0) * D_HEAD;
  const bf16* Kt = kbuf  + (size_t)kv * T_SEQ * D_HEAD;   // [key][d]
  const bf16* Vt = vtbuf + (size_t)kv * D_HEAD * T_SEQ;   // [d][key]

  v16bf qa[4];
#pragma unroll
  for (int c = 0; c < 4; ++c) qa[c] = load_a_frag(Q, D_HEAD, c * 32, lane);

  float mg[8], lg[8], ml[8], ll[8];
  v8f accg[8], accl[8];
#pragma unroll
  for (int i = 0; i < 8; ++i) { mg[i] = NEG_BIG; lg[i] = 0.f; ml[i] = NEG_BIG; ll[i] = 0.f; }
#pragma unroll
  for (int nt = 0; nt < 8; ++nt) { accg[nt] = zero8(); accl[nt] = zero8(); }

  const int i_blk = (q0 / WIN) * WIN;       // tile never straddles a 256-block
  int k_start = i_blk - WIN / 2; if (k_start < 0) k_start = 0;
  const int shift = i_blk - k_start;
  const int upmax = q0 + 15 - shift;

  for (int kb = 0; kb < q0 + 16; kb += 32) {
    // S = Q(16x128) @ K^T(128x32): two 16x16 C-tiles, K accumulated in 4 chunks
    v8f s0 = zero8(), s1 = zero8();
#pragma unroll
    for (int c = 0; c < 4; ++c) {
      v16bf b0 = load_bt_frag(Kt, D_HEAD, kb,      c * 32, lane);
      v16bf b1 = load_bt_frag(Kt, D_HEAD, kb + 16, c * 32, lane);
      s0 = WMMA_BF16(qa[c], b0, s0);
      s1 = WMMA_BF16(qa[c], b1, s1);
    }

    const bool doLocal = (kb + 31 >= k_start) && (kb <= upmax);
    float aG[8], aL[8];
#pragma unroll
    for (int i = 0; i < 8; ++i) {
      const int qrow = q0 + i + 8 * half;   // C-layout row for this vgpr slot
      const int c0 = kb + colb, c1 = c0 + 16;
      // causal
      float g0 = (c0 <= qrow) ? s0[i] : NEG_BIG;
      float g1 = (c1 <= qrow) ? s1[i] : NEG_BIG;
      float p0, p1;
      sm_update(g0, g1, mg[i], lg[i], aG[i], p0, p1);
      ldsP[0][i + 8 * half][colb]      = f2bf(p0);
      ldsP[0][i + 8 * half][colb + 16] = f2bf(p1);
      // local window
      aL[i] = 1.f;
      if (doLocal) {
        const int up = qrow - shift;
        float t0 = (c0 >= k_start && c0 <= up) ? s0[i] : NEG_BIG;
        float t1 = (c1 >= k_start && c1 <= up) ? s1[i] : NEG_BIG;
        float e0, e1;
        sm_update(t0, t1, ml[i], ll[i], aL[i], e0, e1);
        ldsP[1][i + 8 * half][colb]      = f2bf(e0);
        ldsP[1][i + 8 * half][colb + 16] = f2bf(e1);
      }
    }
    __syncthreads();   // single wave: forces s_wait_dscnt before re-reading P in A-layout

    v16bf pG = load_a_frag(&ldsP[0][0][0], 32, 0, lane);
    v16bf pL = pG;
    if (doLocal) pL = load_a_frag(&ldsP[1][0][0], 32, 0, lane);

#pragma unroll
    for (int nt = 0; nt < 8; ++nt) {
      v16bf bv = load_bt_frag(Vt, T_SEQ, nt * 16, kb, lane);  // V tile (32 keys x 16 d)
#pragma unroll
      for (int i = 0; i < 8; ++i) accg[nt][i] *= aG[i];
      accg[nt] = WMMA_BF16(pG, bv, accg[nt]);
      if (doLocal) {
#pragma unroll
        for (int i = 0; i < 8; ++i) accl[nt][i] *= aL[i];
        accl[nt] = WMMA_BF16(pL, bv, accl[nt]);
      }
    }
    __syncthreads();
  }

#pragma unroll
  for (int nt = 0; nt < 8; ++nt) {
#pragma unroll
    for (int i = 0; i < 8; ++i) {
      const int qrow = q0 + i + 8 * half;
      const float o = 0.5f * (accg[nt][i] / lg[i] + accl[nt][i] / ll[i]);
      attn[(size_t)qrow * C_DIM + h * D_HEAD + nt * 16 + colb] = f2bf(o);
    }
  }
}

// ---------------- launcher ----------------

extern "C" void kernel_launch(void* const* d_in, const int* in_sizes, int n_in,
                              void* d_out, int out_size, void* d_ws, size_t ws_size,
                              hipStream_t stream) {
  (void)in_sizes; (void)n_in; (void)out_size; (void)ws_size;
  const float* x  = (const float*)d_in[0];
  const float* wq = (const float*)d_in[1];
  const float* wk = (const float*)d_in[2];
  const float* wv = (const float*)d_in[3];
  const float* wo = (const float*)d_in[4];

  char* ws = (char*)d_ws;
  size_t o = 0;
  bf16*  xb    = (bf16*)(ws + o);  o += (size_t)T_SEQ * C_DIM * 2;      //  8 MB
  bf16*  wqkvT = (bf16*)(ws + o);  o += (size_t)NQKV  * C_DIM * 2;      // 12 MB
  bf16*  woT   = (bf16*)(ws + o);  o += (size_t)C_DIM * C_DIM * 2;      //  8 MB
  float* qkvf  = (float*)(ws + o); o += (size_t)T_SEQ * NQKV  * 4;      // 24 MB
  bf16*  qb    = (bf16*)(ws + o);  o += (size_t)NHEAD * T_SEQ * D_HEAD * 2; // 8 MB
  bf16*  kbuf  = (bf16*)(ws + o);  o += (size_t)NKV   * T_SEQ * D_HEAD * 2; // 2 MB
  bf16*  vtb   = (bf16*)(ws + o);  o += (size_t)NKV   * D_HEAD * T_SEQ * 2; // 2 MB
  bf16*  attnb = (bf16*)(ws + o);  o += (size_t)T_SEQ * C_DIM * 2;      //  8 MB  (~72 MB total)

  {
    int n = T_SEQ * C_DIM;
    cvt_f32_bf16_kernel<<<(n + 255) / 256, 256, 0, stream>>>(x, xb, n);
  }
  {
    int n = NQKV * C_DIM;
    pack_wqkv_kernel<<<(n + 255) / 256, 256, 0, stream>>>(wq, wk, wv, wqkvT);
  }
  {
    int n = C_DIM * C_DIM;
    transpose_wo_kernel<<<(n + 255) / 256, 256, 0, stream>>>(wo, woT);
  }
  // qkv = x @ [wq|wk|wv]   (M=2048, N=3072, K=2048; 128x128 block tiles)
  gemm_bf16_tn<<<dim3(NQKV / 128, T_SEQ / 128), 128, 0, stream>>>(xb, wqkvT, qkvf,
                                                                  T_SEQ, NQKV, C_DIM);
  // RoPE + pack into per-head bf16 layouts
  rope_pack_kernel<<<T_SEQ, 256, 0, stream>>>(qkvf, qb, kbuf, vtb);
  // fused dual-softmax attention
  attn_fused_kernel<<<dim3(T_SEQ / 16, NHEAD), 32, 0, stream>>>(qb, kbuf, vtb, attnb);
  // out = attn @ wo
  gemm_bf16_tn<<<dim3(C_DIM / 128, T_SEQ / 128), 128, 0, stream>>>(attnb, woT, (float*)d_out,
                                                                   T_SEQ, C_DIM, C_DIM);
}
